// GlobalRouters_31035433681146
// MI455X (gfx1250) — matrix operand order repaired and checked by reading, not compiled, not run on hardware
//
#include <hip/hip_runtime.h>
#include <hip/hip_bf16.h>
#include <math.h>

typedef __attribute__((ext_vector_type(16))) __bf16    v16bf;
typedef __attribute__((ext_vector_type(8)))  __bf16    v8bf;
typedef __attribute__((ext_vector_type(8)))  float     v8f;
typedef __attribute__((ext_vector_type(8)))  unsigned  v8u;

#define BS_TOTAL 16384   // B * S
#define SEQ      2048
#define NBATCH   8
#define D_IN     2048
#define DSP      64      // d_space
#define NPOOL    5
#define NNEUR    512

// ---------- helpers ----------
// pack two floats -> two bf16 (round-to-nearest via +0x8000, then one v_perm_b32)
__device__ __forceinline__ unsigned pack2bf(float a, float b) {
  unsigned ua = __builtin_bit_cast(unsigned, a) + 0x8000u;
  unsigned ub = __builtin_bit_cast(unsigned, b) + 0x8000u;
  return __builtin_amdgcn_perm(ub, ua, 0x07060302u);
}

__device__ __forceinline__ __bf16 f2bf1(float f) {
  unsigned u = __builtin_bit_cast(unsigned, f) + 0x8000u;
  return __builtin_bit_cast(__bf16, (unsigned short)(u >> 16));
}

// A-matrix 16x32 bf16 lane pack from f32: elems 0..7 = p[0..7], 8..15 = p[16..23]
__device__ __forceinline__ v16bf packA_f32(const float* __restrict__ p) {
  float4 f0 = *(const float4*)(p +  0);
  float4 f1 = *(const float4*)(p +  4);
  float4 f2 = *(const float4*)(p + 16);
  float4 f3 = *(const float4*)(p + 20);
  v8u u;
  u[0] = pack2bf(f0.x, f0.y); u[1] = pack2bf(f0.z, f0.w);
  u[2] = pack2bf(f1.x, f1.y); u[3] = pack2bf(f1.z, f1.w);
  u[4] = pack2bf(f2.x, f2.y); u[5] = pack2bf(f2.z, f2.w);
  u[6] = pack2bf(f3.x, f3.y); u[7] = pack2bf(f3.z, f3.w);
  return __builtin_bit_cast(v16bf, u);
}

// A operand from a bf16 row: two 16-byte loads, concatenated
__device__ __forceinline__ v16bf loadA_bf16(const __bf16* __restrict__ p) {
  v8bf lo = *(const v8bf*)(p);
  v8bf hi = *(const v8bf*)(p + 16);
  return __builtin_shufflevector(lo, hi, 0,1,2,3,4,5,6,7,8,9,10,11,12,13,14,15);
}

// ---------- kernel 0: zero the weight accumulators ----------
__global__ void zero_kernel(float* __restrict__ w, int n) {
  int i = blockIdx.x * blockDim.x + threadIdx.x;
  if (i < n) w[i] = 0.0f;
}

// ---------- kernel 1a: W[64,2048] f32 -> bf16 (row-major) ----------
__global__ void packW_kernel(const float* __restrict__ W, __bf16* __restrict__ Wbf) {
  int i = blockIdx.x * blockDim.x + threadIdx.x;    // pair index < 65536
  float2 f = ((const float2*)W)[i];
  ((unsigned*)Wbf)[i] = pack2bf(f.x, f.y);
}

// ---------- kernel 1b: L2-normalize embedding rows into ebf[5][512][64] bf16 ----------
__global__ void __launch_bounds__(256) prep_kernel(const float* __restrict__ emb,
                                                   const float* __restrict__ rel,
                                                   __bf16* __restrict__ ebf) {
  int wave = (blockIdx.x * blockDim.x + threadIdx.x) >> 5;
  int lane = threadIdx.x & 31;
  if (wave >= NPOOL * NNEUR) return;
  int p = wave / NNEUR, r = wave % NNEUR;
  // pool order: qk(emb 0:512), v(512:1024), q(1024:1536), k(rel_k), v2(1536:2048)
  const float* src = (p == 3) ? (rel + (size_t)r * DSP)
                              : (emb + ((size_t)((p < 3 ? p : 3) * NNEUR + r)) * DSP);
  float e0 = src[lane], e1 = src[lane + 32];
  float ss = e0 * e0 + e1 * e1;
  #pragma unroll
  for (int off = 16; off; off >>= 1) ss += __shfl_xor(ss, off, 32);
  float inv = 1.0f / fmaxf(sqrtf(ss), 1e-12f);
  __bf16* dst = ebf + (size_t)wave * DSP;
  dst[lane]      = f2bf1(e0 * inv);
  dst[lane + 32] = f2bf1(e1 * inv);
}

// ---------- kernel 2: projection hbf[16384,64] = bf16( x @ W^T + b ) ----------
// block = 8 waves = 2 M-tiles x 4 K-quarters; K-partials merged via ds_add_f32
__global__ void __launch_bounds__(256) proj_kernel(const float* __restrict__ x,
                                                   const __bf16* __restrict__ Wbf,
                                                   const float* __restrict__ bias,
                                                   __bf16* __restrict__ hbf) {
  __shared__ float lacc[2 * 16 * DSP];     // 2 tiles x 16 rows x 64 cols (8 KB)
  int tid = threadIdx.x;
  #pragma unroll
  for (int i = 0; i < 8; i++) lacc[tid + i * 256] = 0.0f;
  __syncthreads();

  int w         = tid >> 5;
  int lane      = tid & 31;
  int l15       = lane & 15;
  int hi        = lane >> 4;
  int tileLocal = w >> 2;                  // 0..1
  int kq        = w & 3;                   // K quarter
  int mt        = blockIdx.x * 2 + tileLocal;
  int abase     = hi ? 8 : 0;              // A lane K sub-base
  int bbase     = hi ? 8 : 0;              // B lane K sub-base

  const float* xr = x + (size_t)(mt * 16 + l15) * D_IN;
  int k0 = kq * (D_IN / 4);

  v8f acc[4] = {};
  for (int kk = k0; kk < k0 + D_IN / 4; kk += 32) {
    __builtin_prefetch(xr + kk + 128, 0, 1);               // global_prefetch_b8
    v16bf A = packA_f32(xr + kk + abase);
    v16bf Bv[4];
    #pragma unroll
    for (int t = 0; t < 4; t++)
      Bv[t] = loadA_bf16(Wbf + (size_t)(t * 16 + l15) * D_IN + kk + bbase);
    #pragma unroll
    for (int t = 0; t < 4; t++)
      acc[t] = __builtin_amdgcn_wmma_f32_16x16x32_bf16(
          false, A, false, Bv[t], (short)0, acc[t], false, false);
  }

  // merge K-partials in LDS
  #pragma unroll
  for (int t = 0; t < 4; t++) {
    int n = t * 16 + l15;
    #pragma unroll
    for (int r = 0; r < 8; r++) {
      int row = hi ? r + 8 : r;
      atomicAdd(&lacc[tileLocal * 1024 + row * DSP + n], acc[t][r]);
    }
  }
  __syncthreads();

  // coalesced epilogue: +bias, convert, store bf16
  #pragma unroll
  for (int i = 0; i < 8; i++) {
    int e   = tid + i * 256;               // 0..2047
    int tl  = e >> 10;
    int row = (e >> 6) & 15;
    int n   = e & 63;
    float v = lacc[e] + bias[n];
    hbf[(size_t)((blockIdx.x * 2 + tl) * 16 + row) * DSP + n] = f2bf1(v);
  }
}

// ---------- kernel 3: fused logits + softmax + importance-weighted accumulate ----------
// grid (1024 s-tiles, 5 pools), block 256 = 8 waves; wave w owns neurons [w*64, w*64+64)
__global__ void __launch_bounds__(256) route_kernel(const __bf16* __restrict__ hbf,
                                                    const __bf16* __restrict__ ebf,
                                                    const float* __restrict__ importance,
                                                    float* __restrict__ weights) {
  __shared__ float sred[8][16];
  __shared__ float rowMax[16];
  __shared__ float rowCoef[16];

  int p       = blockIdx.y;
  int tileRow = blockIdx.x * 16;          // global row in [0, 16384)
  int b       = tileRow / SEQ;
  int w       = threadIdx.x >> 5;
  int lane    = threadIdx.x & 31;
  int l15     = lane & 15;
  int hi      = lane >> 4;
  int abase   = hi ? 8 : 0;
  int bbase   = hi ? 8 : 0;

  const __bf16* hrow = hbf + (size_t)(tileRow + l15) * DSP;
  const __bf16* ep   = ebf + (size_t)p * NNEUR * DSP;

  v8f acc[4] = {};
  #pragma unroll
  for (int kk = 0; kk < DSP; kk += 32) {
    v16bf A = loadA_bf16(hrow + kk + abase);
    v16bf Bv[4];
    #pragma unroll
    for (int t = 0; t < 4; t++) {
      int n = w * 64 + t * 16 + l15;
      Bv[t] = loadA_bf16(ep + (size_t)n * DSP + kk + bbase);
    }
    #pragma unroll
    for (int t = 0; t < 4; t++)
      acc[t] = __builtin_amdgcn_wmma_f32_16x16x32_bf16(
          false, A, false, Bv[t], (short)0, acc[t], false, false);
  }

  // ---- phase 1: per-row max over this wave's 64 neurons, then block-wide ----
  #pragma unroll
  for (int r = 0; r < 8; r++) {
    float m = fmaxf(fmaxf(acc[0][r], acc[1][r]), fmaxf(acc[2][r], acc[3][r]));
    #pragma unroll
    for (int off = 1; off < 16; off <<= 1) m = fmaxf(m, __shfl_xor(m, off, 32));
    if (l15 == 0) sred[w][hi ? r + 8 : r] = m;
  }
  __syncthreads();
  if (threadIdx.x < 16) {
    float m = sred[0][threadIdx.x];
    #pragma unroll
    for (int wv = 1; wv < 8; wv++) m = fmaxf(m, sred[wv][threadIdx.x]);
    rowMax[threadIdx.x] = m;
  }
  __syncthreads();

  // ---- phase 2: exponentiate in place, per-row sums ----
  float rmax[8];
  #pragma unroll
  for (int r = 0; r < 8; r++) rmax[r] = rowMax[hi ? r + 8 : r];
  #pragma unroll
  for (int t = 0; t < 4; t++)
    #pragma unroll
    for (int r = 0; r < 8; r++)
      acc[t][r] = __expf(acc[t][r] - rmax[r]);
  #pragma unroll
  for (int r = 0; r < 8; r++) {
    float s = acc[0][r] + acc[1][r] + acc[2][r] + acc[3][r];
    #pragma unroll
    for (int off = 1; off < 16; off <<= 1) s += __shfl_xor(s, off, 32);
    if (l15 == 0) sred[w][hi ? r + 8 : r] = s;
  }
  __syncthreads();
  if (threadIdx.x < 16) {
    float s = 0.0f;
    #pragma unroll
    for (int wv = 0; wv < 8; wv++) s += sred[wv][threadIdx.x];
    float imp = importance[tileRow + threadIdx.x];   // == [b, s] flattened
    rowCoef[threadIdx.x] = imp / s;
  }
  __syncthreads();

  // ---- phase 3: weights[b,n] += sum_rows coef[row] * prob ----
  float cf[8];
  #pragma unroll
  for (int r = 0; r < 8; r++) cf[r] = rowCoef[hi ? r + 8 : r];
  float* wb = weights + ((size_t)p * NBATCH + b) * NNEUR;
  #pragma unroll
  for (int t = 0; t < 4; t++) {
    float s = 0.0f;
    #pragma unroll
    for (int r = 0; r < 8; r++) s += cf[r] * acc[t][r];
    s += __shfl_xor(s, 16, 32);                      // combine the two row halves
    if (!hi) atomicAdd(&wb[w * 64 + t * 16 + l15], s);
  }
}

// ---------- kernel 4: per (pool,batch) top-k + softmax + index sort ----------
__global__ void __launch_bounds__(256) topk_kernel(const float* __restrict__ weights,
                                                   float* __restrict__ out) {
  __shared__ float sw[512];
  __shared__ float rv[256];
  __shared__ int   ri[256];
  __shared__ float selv[64];
  __shared__ int   seli[64];
  __shared__ int   sorted[64];
  __shared__ float ssum;

  const int kArr[NPOOL]    = {64, 32, 64, 64, 32};
  const int idxOff[NPOOL]  = {0, 1024, 1536, 2560, 3584};
  const int softOff[NPOOL] = {512, 1280, 2048, 3072, 3840};

  int p = blockIdx.x / NBATCH;
  int b = blockIdx.x % NBATCH;
  int k = kArr[p];
  int t = threadIdx.x;

  const float* wb = weights + ((size_t)p * NBATCH + b) * NNEUR;
  sw[t] = wb[t];
  sw[t + 256] = wb[t + 256];
  __syncthreads();

  for (int it = 0; it < k; it++) {
    float v0 = sw[t], v1 = sw[t + 256];
    float v; int i;
    if (v1 > v0) { v = v1; i = t + 256; }
    else { v = v0; i = t; }
    rv[t] = v; ri[t] = i;
    __syncthreads();
    for (int off = 128; off > 0; off >>= 1) {
      if (t < off) {
        float vo = rv[t + off]; int io = ri[t + off];
        if (vo > rv[t] || (vo == rv[t] && io < ri[t])) { rv[t] = vo; ri[t] = io; }
      }
      __syncthreads();
    }
    if (t == 0) {
      selv[it] = rv[0];
      seli[it] = ri[0];
      sw[ri[0]] = -__builtin_inff();
    }
    __syncthreads();
  }

  // softmax over top-k (selection order = descending value, so selv[0] is the max)
  float e = 0.0f;
  if (t < k) { e = __expf(selv[t] - selv[0]); rv[t] = e; }
  __syncthreads();
  if (t == 0) {
    float s = 0.0f;
    for (int j = 0; j < k; j++) s += rv[j];
    ssum = s;
  }
  __syncthreads();

  // sort indices ascending (indices are unique)
  if (t < k) {
    int my = seli[t], rank = 0;
    for (int j = 0; j < k; j++) rank += (seli[j] < my) ? 1 : 0;
    sorted[rank] = my;
  }
  __syncthreads();

  if (t < k) {
    out[idxOff[p]  + b * k + t] = (float)sorted[t];   // indices emitted as floats
    out[softOff[p] + b * k + t] = e / ssum;           // soft stays in top-k order
  }
}

// ---------- launcher ----------
extern "C" void kernel_launch(void* const* d_in, const int* in_sizes, int n_in,
                              void* d_out, int out_size, void* d_ws, size_t ws_size,
                              hipStream_t stream) {
  const float* x    = (const float*)d_in[0];   // [8,2048,2048]
  const float* imp  = (const float*)d_in[1];   // [8,2048]
  const float* W    = (const float*)d_in[2];   // [64,2048]
  const float* bias = (const float*)d_in[3];   // [64]
  const float* emb  = (const float*)d_in[4];   // [2560,64]
  const float* rel  = (const float*)d_in[5];   // [512,64]
  float* out = (float*)d_out;

  __bf16* hbf = (__bf16*)d_ws;                           // 16384*64 bf16 (2 MB)
  __bf16* Wbf = hbf + (size_t)BS_TOTAL * DSP;            // 64*2048 bf16
  __bf16* ebf = Wbf + (size_t)DSP * D_IN;                // 5*512*64 bf16
  float* weights = (float*)(ebf + (size_t)NPOOL * NNEUR * DSP);  // 5*8*512 f32

  hipLaunchKernelGGL(zero_kernel, dim3(80), dim3(256), 0, stream,
                     weights, NPOOL * NBATCH * NNEUR);
  hipLaunchKernelGGL(packW_kernel, dim3(256), dim3(256), 0, stream, W, Wbf);
  hipLaunchKernelGGL(prep_kernel, dim3(320), dim3(256), 0, stream, emb, rel, ebf);
  hipLaunchKernelGGL(proj_kernel, dim3(512), dim3(256), 0, stream, x, Wbf, bias, hbf);
  hipLaunchKernelGGL(route_kernel, dim3(1024, 5), dim3(256), 0, stream,
                     hbf, ebf, imp, weights);
  hipLaunchKernelGGL(topk_kernel, dim3(40), dim3(256), 0, stream, weights, out);
}
